// MultiHeadSelfAttention_17523466568129
// MI455X (gfx1250) — matrix operand order repaired and checked by reading, not compiled, not run on hardware
//
#include <hip/hip_runtime.h>
#include <hip/hip_bf16.h>
#include <stdint.h>

// ---------------- types ----------------
typedef __attribute__((ext_vector_type(16))) __bf16 v16bf;
typedef __attribute__((ext_vector_type(8)))  float  v8f;
typedef __attribute__((ext_vector_type(4)))  unsigned int v4u;

union FragAB { v16bf v; v4u q[2]; };   // 32 bytes: one WMMA bf16 A or B operand per lane

#define B_  2
#define S_  2048
#define E_  1024
#define H_  16
#define D_  64
#define M_  (B_*S_)   // 4096 rows for all projections

// workspace layout (bytes)
static constexpr size_t XBF_OFF = 0;                        // x in bf16:           8 MiB
static constexpr size_t WSW_OFF = 8ull  * 1024 * 1024;      // 4 swizzled weights:  4 x 2 MiB
static constexpr size_t QBF_OFF = 16ull * 1024 * 1024;      // Q, K, Vt bf16:       3 x 8 MiB
static constexpr size_t ATT_OFF = 40ull * 1024 * 1024;      // attn out bf16:       8 MiB
static constexpr size_t WBYTES  = 2ull  * 1024 * 1024;      // one swizzled weight
static constexpr size_t QKVBYTES= 8ull  * 1024 * 1024;      // one of Q/K/Vt

__device__ __forceinline__ unsigned short f2bf(float f) {
  union { float f; unsigned int u; } x; x.f = f;
  unsigned int r = x.u + 0x7FFFu + ((x.u >> 16) & 1u);      // round-to-nearest-even
  return (unsigned short)(r >> 16);
}

__device__ __forceinline__ v8f wmma_bf16(const FragAB& a, const FragAB& b, v8f c) {
  // (neg_a, A, neg_b, B, c_mod, C, reuse_a, reuse_b)
  return __builtin_amdgcn_wmma_f32_16x16x32_bf16(false, a.v, false, b.v, (short)0, c, false, false);
}

// ---------------- conversions ----------------
__global__ void k_convert_x(const float* __restrict__ x, unsigned short* __restrict__ xbf, int n) {
  int i = blockIdx.x * blockDim.x + threadIdx.x;
  int stride = gridDim.x * blockDim.x;
  for (; i < n; i += stride) xbf[i] = f2bf(x[i]);
}

// Swizzle weight W[out=n][in=k] (row-major f32) into B-fragment order:
// frag for (kt,nt): element(lane,i) = Bm[kt*32 + (lane>>4)*16 + i][nt*16 + (lane&15)]
// where Bm[k][n] = W[n][k].
__global__ __launch_bounds__(256) void k_convert_w(const float* w0, const float* w1,
                                                   const float* w2, const float* w3,
                                                   char* __restrict__ ws) {
  const float* W = (blockIdx.z == 0) ? w0 : (blockIdx.z == 1) ? w1 : (blockIdx.z == 2) ? w2 : w3;
  unsigned short* dst = (unsigned short*)(ws + WSW_OFF + (size_t)blockIdx.z * WBYTES);
  const int t    = blockIdx.x * 256 + threadIdx.x;   // 0..65535 (exact)
  const int lane = t & 31;
  const int tile = t >> 5;                           // 0..2047  = kt*64 + nt
  const int kt   = tile >> 6;
  const int nt   = tile & 63;
  const int n    = nt * 16 + (lane & 15);
  const int k0   = kt * 32 + (lane >> 4) * 16;
  unsigned short* d = dst + ((size_t)tile * 32 + lane) * 16;
  const float* src  = W + (size_t)n * E_ + k0;
#pragma unroll
  for (int i = 0; i < 16; ++i) d[i] = f2bf(src[i]);
}

// ---------------- WMMA GEMM core: C[4096,1024] = A[4096,1024] x Bm[1024,1024] ----------------
// A: bf16 row-major. Bsw: bf16 pre-swizzled B fragments. 256 threads = 8 waves (4m x 2n),
// wave tile 32x64 (8 C-fragments -> 0.75 KB moved per WMMA), block tile 128x128, K-steps of 32.
// MODE 0: Q (scale 1/8, bf16 [B,H,S,D]), 1: K (bf16 [B,H,S,D]), 2: V^T (bf16 [B,H,D,S]), 3: f32 row-major.
template <int MODE>
__device__ __forceinline__ void gemm_core(const unsigned short* __restrict__ A,
                                          const unsigned short* __restrict__ Bsw,
                                          void* __restrict__ dst) {
  const int lane = threadIdx.x & 31;
  const int wave = threadIdx.x >> 5;
  const int wm = wave >> 1;            // 0..3
  const int wn = wave & 1;             // 0..1
  const int lm = lane & 15;
  const int hl = lane >> 4;

  const int rbase = blockIdx.x * 128 + wm * 32;
  const int cbase = blockIdx.y * 128 + wn * 64;

  v8f c[2][4] = {};

  for (int kt = 0; kt < 32; ++kt) {
    FragAB a[2], b[4];
#pragma unroll
    for (int mt = 0; mt < 2; ++mt) {
      const v4u* ap = (const v4u*)(A + (size_t)(rbase + mt * 16 + lm) * E_ + kt * 32);
      a[mt].q[0] = ap[2 * hl];        // K = hl*16 + 0..7
      a[mt].q[1] = ap[2 * hl + 1];    // K = hl*16 + 8..15
    }
#pragma unroll
    for (int nt = 0; nt < 4; ++nt) {
      const int tile = kt * (E_ / 16) + (cbase >> 4) + nt;
      const v4u* bp = (const v4u*)(Bsw + ((size_t)tile * 32 + lane) * 16);
      b[nt].q[0] = bp[0];
      b[nt].q[1] = bp[1];
    }
#pragma unroll
    for (int mt = 0; mt < 2; ++mt)
#pragma unroll
      for (int nt = 0; nt < 4; ++nt)
        c[mt][nt] = wmma_bf16(a[mt], b[nt], c[mt][nt]);
  }

#pragma unroll
  for (int mt = 0; mt < 2; ++mt)
#pragma unroll
    for (int nt = 0; nt < 4; ++nt)
#pragma unroll
      for (int j = 0; j < 8; ++j) {
        const int m = rbase + mt * 16 + j + 8 * hl;
        const int n = cbase + nt * 16 + lm;
        float val = c[mt][nt][j];
        if (MODE == 3) {
          ((float*)dst)[(size_t)m * E_ + n] = val;
        } else {
          const int bb = m >> 11, s = m & (S_ - 1);
          const int hh = n >> 6,  d = n & (D_ - 1);
          if (MODE == 0) val *= 0.125f;              // fold 1/sqrt(D) into Q (exact in bf16)
          size_t idx;
          if (MODE == 2) idx = (((size_t)(bb * H_ + hh)) * D_ + d) * S_ + s;   // V^T
          else           idx = (((size_t)(bb * H_ + hh)) * S_ + s) * D_ + d;   // Q, K
          ((unsigned short*)dst)[idx] = f2bf(val);
        }
      }
}

__global__ __launch_bounds__(256) void k_gemm_qkv(const unsigned short* __restrict__ xbf,
                                                  char* __restrict__ ws) {
  const int z = blockIdx.z;
  const unsigned short* Bsw = (const unsigned short*)(ws + WSW_OFF + (size_t)z * WBYTES);
  void* dst = (void*)(ws + QBF_OFF + (size_t)z * QKVBYTES);
  if      (z == 0) gemm_core<0>(xbf, Bsw, dst);
  else if (z == 1) gemm_core<1>(xbf, Bsw, dst);
  else             gemm_core<2>(xbf, Bsw, dst);
}

__global__ __launch_bounds__(256) void k_gemm_out(const unsigned short* __restrict__ attnbf,
                                                  const unsigned short* __restrict__ wosw,
                                                  float* __restrict__ out) {
  gemm_core<3>(attnbf, wosw, out);
}

// ---------------- flash-style causal attention ----------------
// 1 wave handles 16 query rows of one (b,h); 4 waves per block. Keys processed in blocks of 32.
__global__ __launch_bounds__(128) void k_attn(const unsigned short* __restrict__ Q,
                                              const unsigned short* __restrict__ K,
                                              const unsigned short* __restrict__ Vt,
                                              unsigned short* __restrict__ attnOut) {
  __shared__ unsigned short pT[4][16 * 32];   // per-wave P transpose buffer (1 KiB each)

  const int lane = threadIdx.x & 31;
  const int wid  = threadIdx.x >> 5;
  const int lm = lane & 15;
  const int hl = lane >> 4;

  const int idx = blockIdx.x * 4 + wid;       // 0..4095
  const int bh  = idx >> 7;                   // (b*H + h), 0..31
  const int q0  = (idx & 127) << 4;           // query tile base row

  const unsigned short* Qh = Q  + (size_t)bh * S_ * D_;
  const unsigned short* Kh = K  + (size_t)bh * S_ * D_;
  const unsigned short* Vh = Vt + (size_t)bh * D_ * S_;

  // Q A-fragments for d-halves {0,32} (rows fixed for this wave)
  FragAB qa[2];
  {
    const v4u* qrow = (const v4u*)(Qh + (size_t)(q0 + lm) * D_);
#pragma unroll
    for (int dh = 0; dh < 2; ++dh) {
      qa[dh].q[0] = qrow[dh * 4 + 2 * hl];
      qa[dh].q[1] = qrow[dh * 4 + 2 * hl + 1];
    }
  }

  v8f o[4] = {};                 // 16x64 f32 output accumulator
  float mrow[8], lrow[8];
#pragma unroll
  for (int j = 0; j < 8; ++j) { mrow[j] = -3.0e38f; lrow[j] = 0.0f; }

  unsigned short* myP = &pT[wid][0];
  const int nkb = (q0 + 47) >> 5;             // number of 32-key blocks (causal)

  for (int kbi = 0; kbi < nkb; ++kbi) {
    const int kb = kbi << 5;

    // prefetch next key block (K rows are one cacheline each; V is streamed along S)
    if (kbi + 1 < nkb) {
      const int kn = kb + 32;
      __builtin_prefetch(Kh + (size_t)(kn + lane) * D_, 0, 0);
      __builtin_prefetch(Vh + (size_t)lane * S_ + kn, 0, 0);
      __builtin_prefetch(Vh + (size_t)(lane + 32) * S_ + kn, 0, 0);
    }

    // scores = Q @ K^T for 32 keys: two 16x16 fragments, contraction over d in steps of 32
    v8f s0 = {}, s1 = {};
    {
      const v4u* kr0 = (const v4u*)(Kh + (size_t)(kb + lm) * D_);
      const v4u* kr1 = (const v4u*)(Kh + (size_t)(kb + 16 + lm) * D_);
      FragAB kf;
#pragma unroll
      for (int dh = 0; dh < 2; ++dh) {
        kf.q[0] = kr0[dh * 4 + hl * 2];
        kf.q[1] = kr0[dh * 4 + hl * 2 + 1];
        s0 = wmma_bf16(qa[dh], kf, s0);
        kf.q[0] = kr1[dh * 4 + hl * 2];
        kf.q[1] = kr1[dh * 4 + hl * 2 + 1];
        s1 = wmma_bf16(qa[dh], kf, s1);
      }
    }

    // causal mask (wave-uniform branch)
    if (kb + 31 > q0) {
      const int col0 = kb + lm;
#pragma unroll
      for (int j = 0; j < 8; ++j) {
        const int row = q0 + j + 8 * hl;
        if (col0 > row)      s0[j] = -3.0e38f;
        if (col0 + 16 > row) s1[j] = -3.0e38f;
      }
    }

    // online softmax (row j lives across the 16 lanes of this half-wave)
    float p0[8], p1[8];
#pragma unroll
    for (int j = 0; j < 8; ++j) {
      float mx = fmaxf(s0[j], s1[j]);
      mx = fmaxf(mx, __shfl_xor(mx, 1, 16));
      mx = fmaxf(mx, __shfl_xor(mx, 2, 16));
      mx = fmaxf(mx, __shfl_xor(mx, 4, 16));
      mx = fmaxf(mx, __shfl_xor(mx, 8, 16));
      const float mnew = fmaxf(mrow[j], mx);
      const float corr = __expf(mrow[j] - mnew);
      const float e0 = __expf(s0[j] - mnew);
      const float e1 = __expf(s1[j] - mnew);
      float sum = e0 + e1;
      sum += __shfl_xor(sum, 1, 16);
      sum += __shfl_xor(sum, 2, 16);
      sum += __shfl_xor(sum, 4, 16);
      sum += __shfl_xor(sum, 8, 16);
      lrow[j] = lrow[j] * corr + sum;
      mrow[j] = mnew;
      p0[j] = e0; p1[j] = e1;
#pragma unroll
      for (int f = 0; f < 4; ++f) o[f][j] *= corr;
    }

    // transpose P (C-layout -> A-layout) through wave-private LDS
#pragma unroll
    for (int j = 0; j < 8; ++j) {
      myP[(j + 8 * hl) * 32 + lm]      = f2bf(p0[j]);
      myP[(j + 8 * hl) * 32 + lm + 16] = f2bf(p1[j]);
    }
    FragAB pa;
    {
      const v4u* prow = (const v4u*)(myP + lm * 32);
      pa.q[0] = prow[2 * hl];
      pa.q[1] = prow[2 * hl + 1];
    }

    // O += P @ V   (V stored transposed [D,S] -> contiguous B-fragment loads)
#pragma unroll
    for (int f = 0; f < 4; ++f) {
      const v4u* vrow = (const v4u*)(Vh + (size_t)(f * 16 + lm) * S_ + kb + hl * 16);
      FragAB vf;
      vf.q[0] = vrow[0];
      vf.q[1] = vrow[1];
      o[f] = wmma_bf16(pa, vf, o[f]);
    }
  }

  // normalize and write attn output as [B,S,E] bf16 (E index = h*64 + d)
  const int b = bh >> 4, h = bh & 15;
#pragma unroll
  for (int j = 0; j < 8; ++j) {
    const float rl = 1.0f / lrow[j];
    const size_t row = (size_t)b * S_ + q0 + j + 8 * hl;
#pragma unroll
    for (int f = 0; f < 4; ++f)
      attnOut[row * E_ + h * 64 + f * 16 + lm] = f2bf(o[f][j] * rl);
  }
}

// ---------------- launch ----------------
extern "C" void kernel_launch(void* const* d_in, const int* in_sizes, int n_in,
                              void* d_out, int out_size, void* d_ws, size_t ws_size,
                              hipStream_t stream) {
  const float* x  = (const float*)d_in[0];
  const float* wq = (const float*)d_in[1];
  const float* wk = (const float*)d_in[2];
  const float* wv = (const float*)d_in[3];
  const float* wo = (const float*)d_in[4];
  char* ws = (char*)d_ws;

  unsigned short* xbf  = (unsigned short*)(ws + XBF_OFF);
  unsigned short* wosw = (unsigned short*)(ws + WSW_OFF + 3 * WBYTES);
  unsigned short* Qbf  = (unsigned short*)(ws + QBF_OFF);
  unsigned short* Kbf  = (unsigned short*)(ws + QBF_OFF + QKVBYTES);
  unsigned short* Vtbf = (unsigned short*)(ws + QBF_OFF + 2 * QKVBYTES);
  unsigned short* attn = (unsigned short*)(ws + ATT_OFF);

  k_convert_x<<<2048, 256, 0, stream>>>(x, xbf, B_ * S_ * E_);
  k_convert_w<<<dim3(256, 1, 4), 256, 0, stream>>>(wq, wk, wv, wo, ws);
  k_gemm_qkv<<<dim3(32, 8, 3), 256, 0, stream>>>(xbf, ws);
  k_attn<<<1024, 128, 0, stream>>>(Qbf, Kbf, Vtbf, attn);
  k_gemm_out<<<dim3(32, 8, 1), 256, 0, stream>>>(attn, wosw, (float*)d_out);
}
